// GatedAttention_65085934403929
// MI455X (gfx1250) — matrix operand order repaired and checked by reading, not compiled, not run on hardware
//
#include <hip/hip_runtime.h>
#include <hip/hip_bf16.h>
#include <math.h>

// ---------------------------------------------------------------------------
// GatedAttention for MI455X (gfx1250, wave32, WMMA).
//   B=8, C=128, H=W=64, P=4096.
//   Gate conv stack (conv2..4 implicit-GEMM WMMA) -> flash attention with
//   async-LDS double buffering -> 1x1 combiner.
// ---------------------------------------------------------------------------

typedef __attribute__((ext_vector_type(16))) _Float16 v16h;
typedef __attribute__((ext_vector_type(8)))  _Float16 v8h;
typedef __attribute__((ext_vector_type(8)))  float    v8f;

#define CDIM  128
#define PDIM  4096
#define BATCH 8
#define QT    64
#define KT    64

// Build a 16-element f16 fragment from two contiguous 16-byte loads.
__device__ __forceinline__ v16h ld_frag16(const _Float16* p0, const _Float16* p1) {
  v8h lo = *(const v8h*)p0;
  v8h hi = *(const v8h*)p1;
  return __builtin_shufflevector(lo, hi, 0,1,2,3,4,5,6,7,8,9,10,11,12,13,14,15);
}

__device__ __forceinline__ int refl(int i, int n) {
  return (i < 0) ? -i : ((i >= n) ? (2 * n - 2 - i) : i);
}

// CDNA5 async global->LDS copy (16B per lane), tracked by ASYNCcnt.
// LDS byte address = low 32 bits of the generic pointer.
__device__ __forceinline__ void async_copy_b128(void* lds_ptr, const void* gptr) {
  unsigned lds_addr = (unsigned)(size_t)lds_ptr;
  asm volatile("global_load_async_to_lds_b128 %0, %1, off"
               :: "v"(lds_addr), "v"(gptr)
               : "memory");
}
__device__ __forceinline__ void wait_async0() {
  asm volatile("s_wait_asynccnt 0x0" ::: "memory");
}

// ---------------------------------------------------------------------------
// conv1: [8,4,256,256] (imgs||masks) -> [8,32,256,256] f16, k7 p3 s1, relu.
// Direct: Cin=4 is not GEMM-shaped.
// ---------------------------------------------------------------------------
__global__ __launch_bounds__(256) void conv1_kernel(
    const float* __restrict__ imgs, const float* __restrict__ masks,
    const float* __restrict__ w, const float* __restrict__ bias,
    _Float16* __restrict__ out) {
  size_t idx = (size_t)blockIdx.x * 256 + threadIdx.x;   // 2^24 total
  int x = (int)(idx & 255);
  int y = (int)((idx >> 8) & 255);
  int co = (int)((idx >> 16) & 31);
  int b = (int)(idx >> 21);
  float acc = bias[co];
  #pragma unroll
  for (int ci = 0; ci < 4; ++ci) {
    const float* ip = (ci < 3) ? imgs + ((size_t)b * 3 + ci) * 65536
                               : masks + (size_t)b * 65536;
    const float* wp = w + ((size_t)co * 4 + ci) * 49;
    for (int ky = 0; ky < 7; ++ky) {
      int iy = refl(y - 3 + ky, 256);
      const float* irow = ip + (size_t)iy * 256;
      #pragma unroll
      for (int kx = 0; kx < 7; ++kx) {
        int ix = refl(x - 3 + kx, 256);
        acc = fmaf(irow[ix], wp[ky * 7 + kx], acc);
      }
    }
  }
  out[idx] = (_Float16)fmaxf(acc, 0.0f);
}

// ---------------------------------------------------------------------------
// Repack conv weights OIHW f32 -> [ky][kx][co][ci] f16 (tap-major, ci contig).
// ---------------------------------------------------------------------------
__global__ void prepack_kernel(const float* __restrict__ w,
                               _Float16* __restrict__ wp,
                               int Cout, int Cin, int ks) {
  int idx = blockIdx.x * 256 + threadIdx.x;
  int total = Cout * Cin * ks * ks;
  if (idx >= total) return;
  int ci = idx % Cin; int t = idx / Cin;
  int co = t % Cout;  t /= Cout;
  int kx = t % ks;    int ky = t / ks;
  wp[idx] = (_Float16)w[(((size_t)co * Cin + ci) * ks + ky) * ks + kx];
}

// ---------------------------------------------------------------------------
// Implicit-GEMM WMMA conv (reflect pad, relu), f16 in/out, f32 accum.
// Per tap (ky,kx): Out(Cout x NT) += W_tap(Cout x Cin) * In_shift(Cin x NT).
// ---------------------------------------------------------------------------
template <int CIN, int COUT, int KS, int STRIDE, int NT>
__global__ __launch_bounds__(256) void conv_wmma_kernel(
    const _Float16* __restrict__ in, const _Float16* __restrict__ wp,
    const float* __restrict__ bias, _Float16* __restrict__ out,
    int Hin, int Win, int Hout, int Wout, int pad) {
  constexpr int IWT = (NT - 1) * STRIDE + KS;  // staged strip width
  constexpr int NMT = COUT / 16;               // M tiles
  constexpr int NNT = NT / 16;                 // N tiles
  constexpr int T = (NMT * NNT) / 8;           // accum tiles per wave
  static_assert(T >= 1 && (NMT * NNT) % 8 == 0, "tile layout");

  const int b = blockIdx.z;
  const int y = blockIdx.y;
  const int x0 = blockIdx.x * NT;
  const int tid = threadIdx.x;
  const int lane = tid & 31, wave = tid >> 5;
  const int hf = (lane >> 4) & 1, l15 = lane & 15;

  __shared__ __align__(16) _Float16 Ti[KS][IWT][CIN + 8];  // [ky][ix][ci]
  __shared__ int ixmap[IWT];
  __shared__ int iymap[KS];

  if (tid < IWT) ixmap[tid] = refl(x0 * STRIDE - pad + tid, Win);
  if (tid < KS)  iymap[tid] = refl(y * STRIDE - pad + tid, Hin);
  __syncthreads();

  for (int idx = tid; idx < KS * CIN * IWT; idx += 256) {
    int j = idx % IWT;
    int rest = idx / IWT;
    int ci = rest % CIN;
    int ky = rest / CIN;
    Ti[ky][j][ci] =
        in[(((size_t)b * CIN + ci) * Hin + iymap[ky]) * Win + ixmap[j]];
  }
  __syncthreads();

  v8f acc[T];
  #pragma unroll
  for (int j = 0; j < T; ++j) { v8f z = {}; acc[j] = z; }

  #pragma unroll 1
  for (int ky = 0; ky < KS; ++ky) {
    #pragma unroll 1
    for (int kx = 0; kx < KS; ++kx) {
      const _Float16* wtap = wp + (size_t)(ky * KS + kx) * COUT * CIN;
      #pragma unroll
      for (int kc = 0; kc < CIN / 32; ++kc) {
        #pragma unroll
        for (int j = 0; j < T; ++j) {
          int tt = wave + 8 * j;
          int mi = tt % NMT, ni = tt / NMT;
          const _Float16* ap =
              wtap + (size_t)(mi * 16 + l15) * CIN + kc * 32 + hf * 8;
          v16h a = ld_frag16(ap, ap + 16);
          int xx = ni * 16 + l15;
          const _Float16* bp = &Ti[ky][xx * STRIDE + kx][kc * 32 + hf * 8];
          v16h bb = ld_frag16(bp, bp + 16);
          acc[j] = __builtin_amdgcn_wmma_f32_16x16x32_f16(
              false, a, false, bb, (short)0, acc[j], false, false);
        }
      }
    }
  }

  #pragma unroll
  for (int j = 0; j < T; ++j) {
    int tt = wave + 8 * j;
    int mi = tt % NMT, ni = tt / NMT;
    int xg = x0 + ni * 16 + l15;
    #pragma unroll
    for (int r = 0; r < 8; ++r) {
      int co = mi * 16 + hf * 8 + r;
      float v = acc[j][r] + bias[co];
      out[(((size_t)b * COUT + co) * Hout + y) * Wout + xg] =
          (_Float16)fmaxf(v, 0.0f);
    }
  }
}

// ---------------------------------------------------------------------------
// conv5: [8,256,64,64] f16 -> gate [8,4096] f32, k3 p1 s1,
// relu + tan(pi*(tanh(x)-0.5)).
// ---------------------------------------------------------------------------
__global__ __launch_bounds__(256) void conv5_kernel(
    const _Float16* __restrict__ in, const float* __restrict__ w,
    const float* __restrict__ bias, float* __restrict__ gate) {
  int idx = blockIdx.x * 256 + threadIdx.x;   // 8*4096
  if (idx >= BATCH * PDIM) return;
  int x = idx & 63, y = (idx >> 6) & 63, b = idx >> 12;
  float acc = bias[0];
  for (int ci = 0; ci < 256; ++ci) {
    const _Float16* ip = in + ((size_t)b * 256 + ci) * 4096;
    const float* wp = w + ci * 9;
    #pragma unroll
    for (int ky = 0; ky < 3; ++ky) {
      int iy = refl(y - 1 + ky, 64);
      #pragma unroll
      for (int kx = 0; kx < 3; ++kx) {
        int ix = refl(x - 1 + kx, 64);
        acc = fmaf((float)ip[iy * 64 + ix], wp[ky * 3 + kx], acc);
      }
    }
  }
  acc = fmaxf(acc, 0.0f);
  float th = tanhf(acc);
  gate[idx] = tanf(3.1415926f * (th - 0.5f));
}

// ---------------------------------------------------------------------------
// Prep: build f16 operand arrays for the attention GEMMs.
// ---------------------------------------------------------------------------
__global__ __launch_bounds__(256) void prep_kernel(
    const float* __restrict__ feat,
    _Float16* __restrict__ Qh, _Float16* __restrict__ Kh,
    _Float16* __restrict__ Vh) {
  const int b = blockIdx.y;
  const int p0 = blockIdx.x * 64;
  const int tid = threadIdx.x;
  const int x = tid & 63;
  const int g = tid >> 6;

  __shared__ __align__(16) float tile[CDIM][65];
  __shared__ float red[4][64];
  __shared__ float rnorm_s[64];

  for (int i = tid; i < CDIM * 64; i += 256) {
    int c = i >> 6, xx = i & 63;
    tile[c][xx] = feat[((size_t)b * CDIM + c) * PDIM + p0 + xx];
  }
  __syncthreads();

  float ss = 0.0f;
  for (int c = g * 32; c < g * 32 + 32; ++c) {
    float k = tile[c][x] + 1e-7f;
    ss = fmaf(k, k, ss);
  }
  red[g][x] = ss;
  __syncthreads();
  if (tid < 64) {
    float s = red[0][tid] + red[1][tid] + red[2][tid] + red[3][tid];
    rnorm_s[tid] = 1.0f / sqrtf(s);
  }
  __syncthreads();

  float rn = rnorm_s[x];
  for (int c = g * 32; c < g * 32 + 32; ++c) {
    float f = tile[c][x];
    float k = f + 1e-7f;
    size_t rowmaj = ((size_t)b * PDIM + p0 + x) * CDIM + c;
    Qh[rowmaj] = (_Float16)f;
    Kh[rowmaj] = (_Float16)(k * rn);
    Vh[((size_t)b * CDIM + c) * PDIM + p0 + x] = (_Float16)k;
  }
}

// ---------------------------------------------------------------------------
// Flash attention with async-LDS double buffering.
// One batch b and one 64-query tile per workgroup; 256 thr = 8 wave32.
// ---------------------------------------------------------------------------
__device__ __forceinline__ void stage_tiles_async(
    const _Float16* __restrict__ Kh, const _Float16* __restrict__ Vh,
    size_t bP, size_t bC, int k0, int tid,
    _Float16 (*KsD)[CDIM + 8], _Float16 (*VsD)[KT + 8]) {
  #pragma unroll
  for (int u = 0; u < 4; ++u) {               // 64x128 f16 = 1024 x 16B
    int i = tid + u * 256;
    int p = i >> 4, cc = (i & 15) * 8;
    async_copy_b128(&KsD[p][cc], &Kh[(bP + k0 + p) * CDIM + cc]);
  }
  #pragma unroll
  for (int u = 0; u < 4; ++u) {               // 128x64 f16 = 1024 x 16B
    int i = tid + u * 256;
    int c = i >> 3, pp = (i & 7) * 8;
    async_copy_b128(&VsD[c][pp], &Vh[(bC + c) * PDIM + k0 + pp]);
  }
}

__global__ __launch_bounds__(256) void flash_attn_kernel(
    const _Float16* __restrict__ Qh, const _Float16* __restrict__ Kh,
    const _Float16* __restrict__ Vh, const float* __restrict__ gate,
    float* __restrict__ att) {
  const int b   = blockIdx.y;
  const int q0  = blockIdx.x * QT;
  const int tid = threadIdx.x;
  const int lane = tid & 31;
  const int wave = tid >> 5;
  const int hf   = (lane >> 4) & 1;
  const int l15  = lane & 15;
  const size_t bP = (size_t)b * PDIM;
  const size_t bC = (size_t)b * CDIM;

  __shared__ __align__(16) _Float16 Qs[QT][CDIM + 8];      // [x][c]
  __shared__ __align__(16) _Float16 Ks[2][KT][CDIM + 8];   // [buf][p][c] (Kn)
  __shared__ __align__(16) _Float16 VsT[2][CDIM][KT + 8];  // [buf][c][p] (V)
  __shared__ __align__(16) float    Sld[KT][QT + 4];       // [p][x]
  __shared__ __align__(16) _Float16 PT[QT][KT + 8];        // [x][p]
  __shared__ float gts[2][KT];
  __shared__ float red[4][QT];
  __shared__ float mrun[QT], lrun[QT], mnew_s[QT], corr_s[QT];

  // Prologue: async-stage Q tile + first K/V tiles into buffer 0.
  #pragma unroll
  for (int u = 0; u < 4; ++u) {
    int i = tid + u * 256;
    int x = i >> 4, cc = (i & 15) * 8;
    async_copy_b128(&Qs[x][cc], &Qh[(bP + q0 + x) * CDIM + cc]);
  }
  stage_tiles_async(Kh, Vh, bP, bC, 0, tid, Ks[0], VsT[0]);
  if (tid < KT) gts[0][tid] = gate[bP + tid];
  if (tid < QT) { mrun[tid] = -1e30f; lrun[tid] = 0.0f; }
  wait_async0();
  __syncthreads();

  v8f o_acc[4];
  #pragma unroll
  for (int ni = 0; ni < 4; ++ni) { v8f z = {}; o_acc[ni] = z; }

  const int mi  = wave & 3;
  const int ni0 = (wave >> 2) * 2;

  for (int it = 0; it < PDIM / KT; ++it) {
    const int k0 = it * KT;
    const int cur = it & 1, nxt = cur ^ 1;

    // Issue async copies for the next tile while computing this one.
    if (k0 + KT < PDIM) {
      stage_tiles_async(Kh, Vh, bP, bC, k0 + KT, tid, Ks[nxt], VsT[nxt]);
      if (tid < KT) gts[nxt][tid] = gate[bP + k0 + KT + tid];
      if (k0 + 2 * KT < PDIM)  // warm L2 two tiles ahead
        __builtin_prefetch(&Kh[(bP + k0 + 2 * KT + (tid >> 2)) * CDIM], 0, 1);
    }

    // GEMM1: S(64x64) = Kn_tile(64xC) * Q_tile(Cx64)
    v8f s0 = {}; v8f s1 = {};
    #pragma unroll
    for (int kc = 0; kc < 4; ++kc) {
      const _Float16* ap = &Ks[cur][mi * 16 + l15][kc * 32 + hf * 8];
      v16h a = ld_frag16(ap, ap + 16);
      const _Float16* bp0 = &Qs[ni0 * 16 + l15][kc * 32 + hf * 8];
      v16h b0 = ld_frag16(bp0, bp0 + 16);
      const _Float16* bp1 = &Qs[(ni0 + 1) * 16 + l15][kc * 32 + hf * 8];
      v16h b1 = ld_frag16(bp1, bp1 + 16);
      s0 = __builtin_amdgcn_wmma_f32_16x16x32_f16(false, a, false, b0,
                                                  (short)0, s0, false, false);
      s1 = __builtin_amdgcn_wmma_f32_16x16x32_f16(false, a, false, b1,
                                                  (short)0, s1, false, false);
    }
    #pragma unroll
    for (int r = 0; r < 8; ++r) {
      int m = mi * 16 + hf * 8 + r;
      float gv = gts[cur][m];
      Sld[m][ni0 * 16 + l15]       = s0[r] + gv;
      Sld[m][(ni0 + 1) * 16 + l15] = s1[r] + gv;
    }
    __syncthreads();

    // Online softmax over keys (columns of Sld).
    {
      int x = tid & 63, gq = tid >> 6;
      float pm = -1e30f;
      #pragma unroll
      for (int i = 0; i < 16; ++i) pm = fmaxf(pm, Sld[gq * 16 + i][x]);
      red[gq][x] = pm;
      __syncthreads();
      if (tid < QT) {
        float tm = fmaxf(fmaxf(red[0][tid], red[1][tid]),
                         fmaxf(red[2][tid], red[3][tid]));
        float mold = mrun[tid];
        float mn = fmaxf(mold, tm);
        mnew_s[tid] = mn;
        corr_s[tid] = __expf(mold - mn);
        mrun[tid] = mn;
      }
      __syncthreads();
      float mv = mnew_s[x];
      float psum = 0.0f;
      #pragma unroll
      for (int i = 0; i < 16; ++i) {
        float e = __expf(Sld[gq * 16 + i][x] - mv);
        psum += e;
        PT[x][gq * 16 + i] = (_Float16)e;
      }
      red[gq][x] = psum;
      __syncthreads();
      if (tid < QT) {
        float ts = red[0][tid] + red[1][tid] + red[2][tid] + red[3][tid];
        lrun[tid] = lrun[tid] * corr_s[tid] + ts;
      }
      __syncthreads();
    }

    // GEMM2: O(16x64 per wave) += V_tile^T(16xKT) * P(KTx64)
    #pragma unroll
    for (int ni = 0; ni < 4; ++ni) {
      float cr = corr_s[ni * 16 + l15];
      #pragma unroll
      for (int r = 0; r < 8; ++r) o_acc[ni][r] *= cr;
    }
    #pragma unroll
    for (int kk = 0; kk < 2; ++kk) {
      const _Float16* ap = &VsT[cur][wave * 16 + l15][kk * 32 + hf * 8];
      v16h a = ld_frag16(ap, ap + 16);
      #pragma unroll
      for (int ni = 0; ni < 4; ++ni) {
        const _Float16* bp = &PT[ni * 16 + l15][kk * 32 + hf * 8];
        v16h bb = ld_frag16(bp, bp + 16);
        o_acc[ni] = __builtin_amdgcn_wmma_f32_16x16x32_f16(
            false, a, false, bb, (short)0, o_acc[ni], false, false);
      }
    }

    wait_async0();      // next tile landed in LDS (per-wave ASYNCcnt)
    __syncthreads();    // publish across waves; buffers safe to swap
  }

  #pragma unroll
  for (int ni = 0; ni < 4; ++ni) {
    float inv = 1.0f / lrun[ni * 16 + l15];
    int xg = q0 + ni * 16 + l15;
    #pragma unroll
    for (int r = 0; r < 8; ++r) {
      int c = wave * 16 + hf * 8 + r;
      att[(bC + c) * PDIM + xg] = o_acc[ni][r] * inv;
    }
  }
}

// ---------------------------------------------------------------------------
// Combiner: out[b][o][p] = cw[o][0:128].att + cw[o][128:256].feat + cb[o]
// ---------------------------------------------------------------------------
__global__ __launch_bounds__(256) void combiner_kernel(
    const float* __restrict__ att, const float* __restrict__ feat,
    const float* __restrict__ cw, const float* __restrict__ cb,
    float* __restrict__ out) {
  int p = (blockIdx.x & 15) * 256 + threadIdx.x;  // P/256 = 16 tiles
  int o = (blockIdx.x >> 4) & 127;
  int b = blockIdx.x >> 11;
  const float* w0 = cw + (size_t)o * 256;
  const float* ap = att  + (size_t)b * CDIM * PDIM + p;
  const float* fp = feat + (size_t)b * CDIM * PDIM + p;
  float acc = cb[o];
  #pragma unroll 4
  for (int c = 0; c < CDIM; ++c) {
    acc = fmaf(w0[c],        ap[(size_t)c * PDIM], acc);
    acc = fmaf(w0[CDIM + c], fp[(size_t)c * PDIM], acc);
  }
  out[((size_t)b * CDIM + o) * PDIM + p] = acc;
}

// ---------------------------------------------------------------------------
extern "C" void kernel_launch(void* const* d_in, const int* in_sizes, int n_in,
                              void* d_out, int out_size, void* d_ws,
                              size_t ws_size, hipStream_t stream) {
  const float* feat  = (const float*)d_in[0];
  const float* imgs  = (const float*)d_in[1];
  const float* masks = (const float*)d_in[2];
  const float* gw1 = (const float*)d_in[3];  const float* gb1 = (const float*)d_in[4];
  const float* gw2 = (const float*)d_in[5];  const float* gb2 = (const float*)d_in[6];
  const float* gw3 = (const float*)d_in[7];  const float* gb3 = (const float*)d_in[8];
  const float* gw4 = (const float*)d_in[9];  const float* gb4 = (const float*)d_in[10];
  const float* gw5 = (const float*)d_in[11]; const float* gb5 = (const float*)d_in[12];
  const float* cw  = (const float*)d_in[13]; const float* cb  = (const float*)d_in[14];
  float* outp = (float*)d_out;

  const size_t MB = (size_t)1 << 20;
  char* ws = (char*)d_ws;
  _Float16* A1h  = (_Float16*)(ws);             // 32 MiB  [8,32,256,256]
  _Float16* A2h  = (_Float16*)(ws + 32 * MB);   // 16 MiB  [8,64,128,128]
  _Float16* A3h  = (_Float16*)(ws + 48 * MB);   // 32 MiB  [8,128,128,128]
  _Float16* A4h  = (_Float16*)(ws + 80 * MB);   // 16 MiB  [8,256,64,64]
  _Float16* wp2  = (_Float16*)(ws + 96 * MB);   // 100 KiB (pad 1 MiB)
  _Float16* wp3  = (_Float16*)(ws + 97 * MB);   // 400 KiB
  _Float16* wp4  = (_Float16*)(ws + 98 * MB);   // 576 KiB
  float*    gateb= (float*)(ws + 99 * MB);      // 128 KiB
  _Float16* Qh   = (_Float16*)(ws + 100 * MB);  // 8 MiB
  _Float16* Kh   = (_Float16*)(ws + 108 * MB);  // 8 MiB
  _Float16* Vh   = (_Float16*)(ws + 116 * MB);  // 8 MiB
  float*    attb = (float*)(ws + 124 * MB);     // 16 MiB -> 140 MiB total

  auto cdiv = [](size_t a, size_t b) { return (unsigned)((a + b - 1) / b); };

  // ---- Gate conv stack ----
  conv1_kernel<<<cdiv((size_t)8 * 32 * 256 * 256, 256), 256, 0, stream>>>(
      imgs, masks, gw1, gb1, A1h);

  prepack_kernel<<<cdiv((size_t)64 * 32 * 25, 256), 256, 0, stream>>>(
      gw2, wp2, 64, 32, 5);
  prepack_kernel<<<cdiv((size_t)128 * 64 * 25, 256), 256, 0, stream>>>(
      gw3, wp3, 128, 64, 5);
  prepack_kernel<<<cdiv((size_t)256 * 128 * 9, 256), 256, 0, stream>>>(
      gw4, wp4, 256, 128, 3);

  conv_wmma_kernel<32, 64, 5, 2, 64><<<dim3(2, 128, 8), 256, 0, stream>>>(
      A1h, wp2, gb2, A2h, 256, 256, 128, 128, 2);
  conv_wmma_kernel<64, 128, 5, 1, 64><<<dim3(2, 128, 8), 256, 0, stream>>>(
      A2h, wp3, gb3, A3h, 128, 128, 128, 128, 2);
  conv_wmma_kernel<128, 256, 3, 2, 32><<<dim3(2, 64, 8), 256, 0, stream>>>(
      A3h, wp4, gb4, A4h, 128, 128, 64, 64, 1);

  conv5_kernel<<<cdiv((size_t)BATCH * PDIM, 256), 256, 0, stream>>>(
      A4h, gw5, gb5, gateb);

  // ---- Prep f16 operand arrays ----
  prep_kernel<<<dim3(PDIM / 64, BATCH), 256, 0, stream>>>(feat, Qh, Kh, Vh);

  // ---- Flash attention (WMMA + async LDS double buffering) ----
  flash_attn_kernel<<<dim3(PDIM / QT, BATCH), 256, 0, stream>>>(Qh, Kh, Vh,
                                                                gateb, attb);

  // ---- 1x1 combiner ----
  combiner_kernel<<<BATCH * CDIM * (PDIM / 256), 256, 0, stream>>>(
      attb, feat, cw, cb, outp);
}
